// _ATTR_NETWORK_59992103190650
// MI455X (gfx1250) — matrix operand order repaired and checked by reading, not compiled, not run on hardware
//
#include <hip/hip_runtime.h>
#include <hip/hip_bf16.h>
#include <math.h>

// ---------------- problem constants ----------------
#define BB    256
#define LL    200
#define VV    50000
#define DD    256
#define HH    4
#define DHD   64
#define DFF   2048
#define TT    (BB*LL)          // 51200 tokens
#define LP    208              // L padded to multiple of 16
#define NQT   13               // ceil(200/16)
#define SCALE 0.125f           // 1/sqrt(64)
#define EPSL  1e-5f

typedef __attribute__((ext_vector_type(16))) _Float16 v16h;
typedef __attribute__((ext_vector_type(8)))  float    v8f;

#define WMMA16(c, a, b) \
  (c) = __builtin_amdgcn_wmma_f32_16x16x32_f16(false, (a), false, (b), (short)0, (c), false, false)

// ---------------- fragment loaders (wave32; contiguous k order, same for A and B) ----------------
// A 16x32 f16: lane m = lane&15, K(j) = (lane>>4)*16 + j
__device__ __forceinline__ v16h frag_a_lds(const _Float16* base, int stride, int k0, int lane) {
  int m  = lane & 15;
  int kb = (lane >> 4) << 4;
  const _Float16* p = base + m * stride + k0 + kb;
  v16h a;
#pragma unroll
  for (int j = 0; j < 16; ++j) a[j] = p[j];
  return a;
}

// B 32x16 f16: lane n = lane&15, K(j) = (lane>>4)*16 + j
// source laid out as Bp[n][k] (Y = A @ W^T, W row-major [N,K]) -> 16 contiguous halves per lane
__device__ __forceinline__ v16h frag_b_nk_f16(const _Float16* __restrict__ Bp, int ldb,
                                              int n0, int k0, int lane) {
  int n  = lane & 15;
  int kb = (lane >> 4) << 4;
  const _Float16* p = Bp + (size_t)(n0 + n) * ldb + k0 + kb;
  v16h b;
#pragma unroll
  for (int j = 0; j < 16; ++j) b[j] = p[j];
  return b;
}

// B laid out as Bp[k][n] (row index = K), f16 source (used for P @ V)
__device__ __forceinline__ v16h frag_b_kn_f16(const _Float16* __restrict__ Bp, int ldb,
                                              int n0, int k0, int lane) {
  int n  = lane & 15;
  int kb = (lane >> 4) << 4;
  v16h b;
#pragma unroll
  for (int j = 0; j < 16; ++j) b[j] = Bp[(size_t)(k0 + kb + j) * ldb + n0 + n];
  return b;
}

__device__ __forceinline__ v8f acc_fill(float v) {
  v8f c;
#pragma unroll
  for (int r = 0; r < 8; ++r) c[r] = v;
  return c;
}

// 8-lane (row-group) tree reduction within a wave32
__device__ __forceinline__ float red8(float v) {
  v += __shfl_xor(v, 1, 32);
  v += __shfl_xor(v, 2, 32);
  v += __shfl_xor(v, 4, 32);
  return v;
}
__device__ __forceinline__ float redmax8(float v) {
  v = fmaxf(v, __shfl_xor(v, 1, 32));
  v = fmaxf(v, __shfl_xor(v, 2, 32));
  v = fmaxf(v, __shfl_xor(v, 4, 32));
  return v;
}

// ---------------- small utility kernels ----------------
__global__ void k_zero(float* p, int n) {
  int i = blockIdx.x * 256 + threadIdx.x;
  if (i < n) p[i] = 0.0f;
}

// one-shot f32 -> f16 weight conversion (weights stay in L2 afterwards)
__global__ void k_cvt(const float* __restrict__ src, _Float16* __restrict__ dst, int n) {
  int i = blockIdx.x * 256 + threadIdx.x;
  if (i * 4 < n) {
    float4 v4 = ((const float4*)src)[i];
    _Float16* d = dst + i * 4;
    d[0] = (_Float16)v4.x; d[1] = (_Float16)v4.y;
    d[2] = (_Float16)v4.z; d[3] = (_Float16)v4.w;
  }
}

__global__ void k_masks(const int* __restrict__ attr_lens, const int* __restrict__ lens,
                        float* __restrict__ pad_out, float* __restrict__ mask_out) {
  int t = blockIdx.x * 256 + threadIdx.x;
  if (t < TT) {
    int b = t / LL, l = t % LL;
    pad_out[t]  = (l >= attr_lens[b]) ? 1.0f : 0.0f;
    mask_out[t] = (l >= lens[b])      ? 1.0f : 0.0f;
  }
}

__global__ void k_gather(const int* __restrict__ ids, const float* __restrict__ emb,
                         float* __restrict__ x) {
  int t = blockIdx.x, d = threadIdx.x;
  x[(size_t)t * DD + d] = emb[(size_t)ids[t] * DD + d];
}

// zero the padded rows (l = 200..207) of q/k/v so WMMA tiles never see garbage
__global__ void k_zpad(_Float16* qb, _Float16* kb, _Float16* vb) {
  int i = blockIdx.x * 256 + threadIdx.x;           // B*H*8*64 = 524288
  if (i < BB * HH * 8 * DHD) {
    int bh = i / (8 * DHD);
    int rem = i % (8 * DHD);
    int l = LL + rem / DHD;
    int d = rem % DHD;
    size_t off = ((size_t)bh * LP + l) * DHD + d;
    qb[off] = (_Float16)0.0f;
    kb[off] = (_Float16)0.0f;
    vb[off] = (_Float16)0.0f;
  }
}

// u[b] = user_emb[user_ids[b]] @ user_w^T + user_b   (tiny: scalar kernel)
__global__ void k_user(const int* __restrict__ uid, const float* __restrict__ uemb,
                       const float* __restrict__ uw, const float* __restrict__ ub,
                       float* __restrict__ u) {
  int b = blockIdx.x, j = threadIdx.x;
  const float* e = uemb + (size_t)uid[b] * DD;
  const float* w = uw + (size_t)j * DD;
  float s = ub[j];
#pragma unroll 4
  for (int d = 0; d < DD; ++d) s += e[d] * w[d];
  u[b * DD + j] = s;
}

// vectorized f32 activation tile -> f16 LDS tile (float4 loads, 4-half stores)
__device__ __forceinline__ void stage_tile(const float* __restrict__ src, int t0,
                                           _Float16* dstLds, int stride, int tid) {
  for (int e = tid; e < 16 * (DD / 4); e += 128) {
    int r = e >> 6, c4 = e & 63;
    float4 v4 = ((const float4*)src)[(size_t)(t0 + r) * (DD / 4) + c4];
    _Float16* d = dstLds + r * stride + c4 * 4;
    d[0] = (_Float16)v4.x; d[1] = (_Float16)v4.y;
    d[2] = (_Float16)v4.z; d[3] = (_Float16)v4.w;
  }
}

#define XS 272

// residual add + LayerNorm over a 16x256 f32 LDS tile, 8 threads per row
__device__ __forceinline__ void resid_ln(float* Ys, float* __restrict__ x, int t0, int tid,
                                         const float* __restrict__ g,
                                         const float* __restrict__ be) {
  int row = tid >> 3, sub = tid & 7;
  float* Yr = Ys + row * XS;
  float* xr = x + (size_t)(t0 + row) * DD;
  float s = 0.0f;
  for (int c = sub; c < DD; c += 8) { float y = Yr[c] + xr[c]; Yr[c] = y; s += y; }
  float mu = red8(s) * (1.0f / DD);
  float v = 0.0f;
  for (int c = sub; c < DD; c += 8) { float dv = Yr[c] - mu; v += dv * dv; }
  float inv = rsqrtf(red8(v) * (1.0f / DD) + EPSL);
  for (int c = sub; c < DD; c += 8) xr[c] = (Yr[c] - mu) * inv * g[c] + be[c];
}

// ---------------- QKV projection: qkv = x @ Wqkv^T + bqkv ----------------
#define QKS 784   // 768 + pad (keeps 16B alignment)
__global__ void k_qkv(const float* __restrict__ x, const _Float16* __restrict__ Wh,
                      const float* __restrict__ bqkv,
                      _Float16* __restrict__ qb, _Float16* __restrict__ kbuf,
                      _Float16* __restrict__ vb) {
  __shared__ _Float16 Xs[16 * XS];
  __shared__ _Float16 Ys[16 * QKS];
  int t0 = blockIdx.x * 16;
  int tid = threadIdx.x, lane = tid & 31, w = tid >> 5;
  stage_tile(x, t0, Xs, XS, tid);
  __syncthreads();
  // two independent accumulator chains per wave, shared A fragment
  for (int nt = w; nt < 48; nt += 8) {
    int n0 = nt * 16, n1 = (nt + 4) * 16;
    v8f acc0 = acc_fill(bqkv[n0 + (lane & 15)]);
    v8f acc1 = acc_fill(bqkv[n1 + (lane & 15)]);
#pragma unroll
    for (int kc = 0; kc < 8; ++kc) {
      v16h a = frag_a_lds(Xs, XS, kc * 32, lane);
      WMMA16(acc0, a, frag_b_nk_f16(Wh, DD, n0, kc * 32, lane));
      WMMA16(acc1, a, frag_b_nk_f16(Wh, DD, n1, kc * 32, lane));
    }
    int n = lane & 15;
#pragma unroll
    for (int r = 0; r < 8; ++r) {
      int m = r + ((lane >> 4) << 3);
      Ys[m * QKS + n0 + n] = (_Float16)acc0[r];
      Ys[m * QKS + n1 + n] = (_Float16)acc1[r];
    }
  }
  __syncthreads();
  // coalesced 16B scatter into q/k/v [bh][l][dh]
  for (int cidx = tid; cidx < 16 * 96; cidx += 128) {
    int m = cidx / 96, col0 = (cidx % 96) * 8;
    int sec = col0 >> 8, jj = col0 & 255, h = jj >> 6, dh = jj & 63;
    int t = t0 + m, bi = t / LL, l = t % LL;
    _Float16* dst = (sec == 0) ? qb : (sec == 1) ? kbuf : vb;
    uint4 v4 = *(const uint4*)(Ys + m * QKS + col0);
    *(uint4*)(dst + (((size_t)(bi * HH + h)) * LP + l) * DHD + dh) = v4;
  }
}

// ---------------- fused attention per (b,h,qtile) ----------------
#define QSS 80
__global__ void k_attn(const _Float16* __restrict__ qb, const _Float16* __restrict__ kbuf,
                       const _Float16* __restrict__ vb, const int* __restrict__ attr_lens,
                       float* __restrict__ ob) {
  __shared__ _Float16 Qs[16 * QSS];
  __shared__ float    Ss[16 * 224];
  __shared__ _Float16 Ps[16 * 240];
  int bh = blockIdx.x / NQT, qt = blockIdx.x % NQT;
  int b = bh >> 2, h = bh & 3;
  int tid = threadIdx.x, lane = tid & 31, w = tid >> 5;
  const _Float16* qp = qb   + (size_t)bh * LP * DHD;
  const _Float16* kp = kbuf + (size_t)bh * LP * DHD;
  const _Float16* vp = vb   + (size_t)bh * LP * DHD;
  {  // 16x64 f16 tile copy as aligned 16B chunks: 128 chunks == 1 per thread
    int r = tid >> 3, c8 = tid & 7;
    uint4 v4 = ((const uint4*)qp)[(size_t)(qt * 16 + r) * (DHD / 8) + c8];
    *((uint4*)(Qs + r * QSS + c8 * 8)) = v4;
  }
  __syncthreads();
  // phase 1: S = (Q @ K^T) * scale  (Q fragments hoisted)
  {
    v16h a0 = frag_a_lds(Qs, QSS, 0, lane);
    v16h a1 = frag_a_lds(Qs, QSS, 32, lane);
    for (int kt = w; kt < NQT; kt += 4) {
      v8f acc = acc_fill(0.0f);
      WMMA16(acc, a0, frag_b_nk_f16(kp, DHD, kt * 16, 0, lane));
      WMMA16(acc, a1, frag_b_nk_f16(kp, DHD, kt * 16, 32, lane));
      int n = lane & 15;
#pragma unroll
      for (int r = 0; r < 8; ++r) {
        int m = r + ((lane >> 4) << 3);
        Ss[m * 224 + kt * 16 + n] = acc[r] * SCALE;
      }
    }
  }
  __syncthreads();
  // masked softmax: 8 threads per row, shuffle reductions
  {
    int row = tid >> 3, sub = tid & 7;
    int kl = attr_lens[b]; if (kl > LL) kl = LL;
    float* Sr = Ss + row * 224;
    float mx = -1e30f;
    for (int c = sub; c < kl; c += 8) mx = fmaxf(mx, Sr[c]);
    mx = redmax8(mx);
    float sum = 0.0f;
    for (int c = sub; c < kl; c += 8) { float e = __expf(Sr[c] - mx); Sr[c] = e; sum += e; }
    sum = red8(sum);
    float inv = 1.0f / sum;
    _Float16* Pr = Ps + row * 240;
    for (int c = sub; c < 224; c += 8) Pr[c] = (c < kl) ? (_Float16)(Sr[c] * inv) : (_Float16)0.0f;
  }
  __syncthreads();
  // phase 2: O = P @ V  (each wave one 16-col tile of DH; split accumulators for ILP)
  {
    int n0 = w * 16;
    v8f acc0 = acc_fill(0.0f);
    v8f acc1 = acc_fill(0.0f);
#pragma unroll
    for (int kc = 0; kc < 6; kc += 2) {
      WMMA16(acc0, frag_a_lds(Ps, 240, kc * 32, lane),
             frag_b_kn_f16(vp, DHD, n0, kc * 32, lane));
      WMMA16(acc1, frag_a_lds(Ps, 240, (kc + 1) * 32, lane),
             frag_b_kn_f16(vp, DHD, n0, (kc + 1) * 32, lane));
    }
    WMMA16(acc0, frag_a_lds(Ps, 240, 192, lane), frag_b_kn_f16(vp, DHD, n0, 192, lane));
    int n = lane & 15;
#pragma unroll
    for (int r = 0; r < 8; ++r) {
      int m = r + ((lane >> 4) << 3);
      int l = qt * 16 + m;
      if (l < LL) {
        int t = b * LL + l;
        ob[(size_t)t * DD + h * DHD + n0 + n] = acc0[r] + acc1[r];
      }
    }
  }
}

// ---------------- Wo projection + residual + LayerNorm1 ----------------
__global__ void k_proj_ln(const float* __restrict__ ob, float* __restrict__ x,
                          const _Float16* __restrict__ Woh, const float* __restrict__ bo,
                          const float* __restrict__ g, const float* __restrict__ be) {
  __shared__ _Float16 As[16 * XS];
  __shared__ float    Ys[16 * XS];
  int t0 = blockIdx.x * 16;
  int tid = threadIdx.x, lane = tid & 31, w = tid >> 5;
  stage_tile(ob, t0, As, XS, tid);
  __syncthreads();
  {
    v8f acc[4];
#pragma unroll
    for (int i = 0; i < 4; ++i) acc[i] = acc_fill(bo[(w + 4 * i) * 16 + (lane & 15)]);
#pragma unroll 1
    for (int kc = 0; kc < 8; ++kc) {
      v16h a = frag_a_lds(As, XS, kc * 32, lane);
#pragma unroll
      for (int i = 0; i < 4; ++i)
        WMMA16(acc[i], a, frag_b_nk_f16(Woh, DD, (w + 4 * i) * 16, kc * 32, lane));
    }
    int n = lane & 15;
#pragma unroll
    for (int i = 0; i < 4; ++i)
#pragma unroll
      for (int r = 0; r < 8; ++r) {
        int m = r + ((lane >> 4) << 3);
        Ys[m * XS + (w + 4 * i) * 16 + n] = acc[i][r];
      }
  }
  __syncthreads();
  resid_ln(Ys, x, t0, tid, g, be);
}

// ---------------- fused FFN (chunked over DFF) + residual + LayerNorm2 ----------------
#define HS 528   // 512-chunk + pad
__global__ void k_ffn(float* __restrict__ x, const _Float16* __restrict__ W1h,
                      const float* __restrict__ b1, const _Float16* __restrict__ W2h,
                      const float* __restrict__ b2, const float* __restrict__ g,
                      const float* __restrict__ be) {
  __shared__ _Float16 Xs[16 * XS];
  __shared__ _Float16 Hs[16 * HS];
  __shared__ float    Ys[16 * XS];
  int t0 = blockIdx.x * 16;
  int tid = threadIdx.x, lane = tid & 31, w = tid >> 5;
  stage_tile(x, t0, Xs, XS, tid);
  __syncthreads();
  v8f acc[4];
#pragma unroll
  for (int i = 0; i < 4; ++i) acc[i] = acc_fill(b2[(w + 4 * i) * 16 + (lane & 15)]);

#pragma unroll 1
  for (int ch = 0; ch < 4; ++ch) {
    int kbase = ch * 512;
    // warm L2/L0 for the phase-2 weights of this chunk while phase 1 runs
    __builtin_prefetch(W2h + (size_t)(w * 4) * 16 * DFF + kbase, 0, 1);
    // phase 1: H = relu(X @ W1^T + b1); two chains sharing the A fragment
#pragma unroll 1
    for (int nt = w; nt < 32; nt += 8) {
      int n0g = kbase + nt * 16, n1g = kbase + (nt + 4) * 16;
      v8f h0 = acc_fill(b1[n0g + (lane & 15)]);
      v8f h1 = acc_fill(b1[n1g + (lane & 15)]);
#pragma unroll
      for (int kc = 0; kc < 8; ++kc) {
        v16h a = frag_a_lds(Xs, XS, kc * 32, lane);
        WMMA16(h0, a, frag_b_nk_f16(W1h, DD, n0g, kc * 32, lane));
        WMMA16(h1, a, frag_b_nk_f16(W1h, DD, n1g, kc * 32, lane));
      }
      int n = lane & 15;
#pragma unroll
      for (int r = 0; r < 8; ++r) {
        int m = r + ((lane >> 4) << 3);
        Hs[m * HS + (nt - w) * 16 + w * 16 + n]       = (_Float16)fmaxf(h0[r], 0.0f);
        Hs[m * HS + (nt + 4 - w) * 16 + w * 16 + n]   = (_Float16)fmaxf(h1[r], 0.0f);
      }
    }
    __syncthreads();
    // phase 2: Y += H_chunk @ W2_chunk^T; 4 chains sharing A
#pragma unroll 1
    for (int kc = 0; kc < 16; ++kc) {
      v16h a = frag_a_lds(Hs, HS, kc * 32, lane);
#pragma unroll
      for (int i = 0; i < 4; ++i)
        WMMA16(acc[i], a, frag_b_nk_f16(W2h, DFF, (w + 4 * i) * 16, kbase + kc * 32, lane));
    }
    __syncthreads();
  }
  {
    int n = lane & 15;
#pragma unroll
    for (int i = 0; i < 4; ++i)
#pragma unroll
      for (int r = 0; r < 8; ++r) {
        int m = r + ((lane >> 4) << 3);
        Ys[m * XS + (w + 4 * i) * 16 + n] = acc[i][r];
      }
  }
  __syncthreads();
  resid_ln(Ys, x, t0, tid, g, be);
}

// ---------------- final head: attr_h GEMM + dot(u) + scatter-add ----------------
__global__ void k_final(const float* __restrict__ x, const _Float16* __restrict__ attr_wh,
                        const float* __restrict__ attr_b, const float* __restrict__ u,
                        const float* __restrict__ tf, const int* __restrict__ attr_lens,
                        const int* __restrict__ idx, const float* __restrict__ aiw,
                        const float* __restrict__ aib, float* __restrict__ ual_out,
                        float* __restrict__ logits) {
  __shared__ _Float16 As[16 * XS];
  __shared__ float    Ys[16 * XS];
  int t0 = blockIdx.x * 16;
  int tid = threadIdx.x, lane = tid & 31, w = tid >> 5;
  stage_tile(x, t0, As, XS, tid);
  __syncthreads();
  {
    v8f acc[4];
#pragma unroll
    for (int i = 0; i < 4; ++i) acc[i] = acc_fill(attr_b[(w + 4 * i) * 16 + (lane & 15)]);
#pragma unroll 1
    for (int kc = 0; kc < 8; ++kc) {
      v16h a = frag_a_lds(As, XS, kc * 32, lane);
#pragma unroll
      for (int i = 0; i < 4; ++i)
        WMMA16(acc[i], a, frag_b_nk_f16(attr_wh, DD, (w + 4 * i) * 16, kc * 32, lane));
    }
    int n = lane & 15;
#pragma unroll
    for (int i = 0; i < 4; ++i)
#pragma unroll
      for (int r = 0; r < 8; ++r) {
        int m = r + ((lane >> 4) << 3);
        Ys[m * XS + (w + 4 * i) * 16 + n] = acc[i][r];
      }
  }
  __syncthreads();
  {
    int row = tid >> 3, sub = tid & 7;
    int t = t0 + row, bi = t / LL, l = t % LL;
    const float* ur = u + (size_t)bi * DD;
    float s = 0.0f;
    for (int c = sub; c < DD; c += 8) s += Ys[row * XS + c] * ur[c];
    s = red8(s);
    if (sub == 0) {
      float v = s + aiw[0] * tf[t] + aib[0];
      ual_out[t] = v;
      if (l < attr_lens[bi]) atomicAdd(&logits[(size_t)bi * VV + idx[t]], v);
    }
  }
}

// ---------------- launcher ----------------
extern "C" void kernel_launch(void* const* d_in, const int* in_sizes, int n_in,
                              void* d_out, int out_size, void* d_ws, size_t ws_size,
                              hipStream_t stream) {
  // inputs (setup_inputs order)
  const int*   attr_item      = (const int*)  d_in[0];
  const float* attr_tf_item   = (const float*)d_in[1];
  const int*   attr_lens_item = (const int*)  d_in[2];
  const int*   attr_index     = (const int*)  d_in[3];
  const int*   user_ids       = (const int*)  d_in[9];
  const int*   lens           = (const int*)  d_in[10];
  const float* attr_emb       = (const float*)d_in[12];
  const float* user_emb       = (const float*)d_in[13];
  const float* Wqkv           = (const float*)d_in[14];
  const float* bqkv           = (const float*)d_in[15];
  const float* Wo             = (const float*)d_in[16];
  const float* bo             = (const float*)d_in[17];
  const float* ln1_g          = (const float*)d_in[18];
  const float* ln1_b          = (const float*)d_in[19];
  const float* W1             = (const float*)d_in[20];
  const float* b1             = (const float*)d_in[21];
  const float* W2             = (const float*)d_in[22];
  const float* b2             = (const float*)d_in[23];
  const float* ln2_g          = (const float*)d_in[24];
  const float* ln2_b          = (const float*)d_in[25];
  const float* user_w         = (const float*)d_in[26];
  const float* user_b         = (const float*)d_in[27];
  const float* attr_w         = (const float*)d_in[28];
  const float* attr_b         = (const float*)d_in[29];
  const float* ai_w           = (const float*)d_in[30];
  const float* ai_b           = (const float*)d_in[31];

  // outputs: (ual[B*L], pad[B*L], logits[B*V], mask[B*L]) as f32
  float* out      = (float*)d_out;
  float* ual_out  = out;
  float* pad_out  = out + TT;
  float* log_out  = out + 2 * TT;
  float* mask_out = out + 2 * TT + (size_t)BB * VV;

  // workspace carve-up
  float*     x   = (float*)d_ws;                       // [T, D] f32
  float*     ob  = x + (size_t)TT * DD;                // [T, D] f32
  float*     u   = ob + (size_t)TT * DD;               // [B, D] f32
  _Float16*  qb  = (_Float16*)(u + BB * DD);           // [B*H, LP, DH] f16
  _Float16*  kb  = qb + (size_t)BB * HH * LP * DHD;
  _Float16*  vb  = kb + (size_t)BB * HH * LP * DHD;
  _Float16*  wqh = vb + (size_t)BB * HH * LP * DHD;    // f16 weight copies
  _Float16*  woh = wqh + (size_t)2 * 3 * DD * DD;      //  Wqkv  2*768*256
  _Float16*  w1h = woh + (size_t)2 * DD * DD;          //  Wo    2*256*256
  _Float16*  w2h = w1h + (size_t)2 * DFF * DD;         //  W1    2*2048*256
  _Float16*  awh = w2h + (size_t)2 * DD * DFF;         //  W2    2*256*2048
                                                       //  attr_w  256*256

  const int TILES = TT / 16;                           // 3200

  // one-shot weight conversion to f16 (served from L2 afterwards)
  {
    int n;
    n = 2 * 3 * DD * DD; k_cvt<<<dim3((n / 4 + 255) / 256), dim3(256), 0, stream>>>(Wqkv,   wqh, n);
    n = 2 * DD * DD;     k_cvt<<<dim3((n / 4 + 255) / 256), dim3(256), 0, stream>>>(Wo,     woh, n);
    n = 2 * DFF * DD;    k_cvt<<<dim3((n / 4 + 255) / 256), dim3(256), 0, stream>>>(W1,     w1h, n);
    n = 2 * DD * DFF;    k_cvt<<<dim3((n / 4 + 255) / 256), dim3(256), 0, stream>>>(W2,     w2h, n);
    n = DD * DD;         k_cvt<<<dim3((n / 4 + 255) / 256), dim3(256), 0, stream>>>(attr_w, awh, n);
  }

  k_zero<<<dim3((BB * VV + 255) / 256), dim3(256), 0, stream>>>(log_out, BB * VV);
  k_masks<<<dim3((TT + 255) / 256), dim3(256), 0, stream>>>(attr_lens_item, lens, pad_out, mask_out);
  k_gather<<<dim3(TT), dim3(DD), 0, stream>>>(attr_item, attr_emb, x);

  for (int layer = 0; layer < 2; ++layer) {
    k_qkv<<<dim3(TILES), dim3(128), 0, stream>>>(x, wqh + (size_t)layer * 3 * DD * DD,
                                                 bqkv + (size_t)layer * 3 * DD, qb, kb, vb);
    k_zpad<<<dim3((BB * HH * 8 * DHD + 255) / 256), dim3(256), 0, stream>>>(qb, kb, vb);
    k_attn<<<dim3(BB * HH * NQT), dim3(128), 0, stream>>>(qb, kb, vb, attr_lens_item, ob);
    k_proj_ln<<<dim3(TILES), dim3(128), 0, stream>>>(ob, x,
        woh + (size_t)layer * DD * DD, bo + (size_t)layer * DD,
        ln1_g + (size_t)layer * DD, ln1_b + (size_t)layer * DD);
    k_ffn<<<dim3(TILES), dim3(128), 0, stream>>>(x,
        w1h + (size_t)layer * DFF * DD, b1 + (size_t)layer * DFF,
        w2h + (size_t)layer * DD * DFF, b2 + (size_t)layer * DD,
        ln2_g + (size_t)layer * DD, ln2_b + (size_t)layer * DD);
  }

  k_user<<<dim3(BB), dim3(DD), 0, stream>>>(user_ids, user_emb, user_w, user_b, u);
  k_final<<<dim3(TILES), dim3(128), 0, stream>>>(x, awh, attr_b, u, attr_tf_item,
                                                 attr_lens_item, attr_index, ai_w, ai_b,
                                                 ual_out, log_out);
}
